// CustomTransformerEncoderLayer_32667521254107
// MI455X (gfx1250) — matrix operand order repaired and checked by reading, compiled, not run on hardware
//
#include <hip/hip_runtime.h>
#include <math.h>

// ---- problem constants (match reference) ----
static constexpr int E  = 1024;
static constexpr int Hh = 16;
static constexpr int DH = 64;
static constexpr int Ff = 4096;
static constexpr int Bb = 4;
static constexpr int Ss = 1024;
static constexpr float EPS = 1e-5f;

typedef __attribute__((ext_vector_type(16))) _Float16 v16h;
typedef __attribute__((ext_vector_type(8)))  _Float16 v8h;
typedef __attribute__((ext_vector_type(8)))  float    v8f;

// ------------------------------------------------------------------
// Fragment loaders (16x16x32 f16 WMMA; contiguous k order chosen by this file)
// A 16x32: lane holds row (lane&15); slot e -> k = half*16+e (contiguous),
//          half = lane>>4.
// B 32x16: lane holds col (lane&15); slot e -> k = half*16+e (contiguous!).
// C/D v8f: row = j + 8*half, col = lane&15.
// ------------------------------------------------------------------
__device__ inline v16h load_a16(const _Float16* __restrict__ rowp, int half) {
  const v8h lo = *(const v8h*)(rowp + half * 16);
  const v8h hi = *(const v8h*)(rowp + half * 16 + 8);
  return __builtin_shufflevector(lo, hi, 0,1,2,3,4,5,6,7,8,9,10,11,12,13,14,15);
}

__device__ inline v16h load_b16(const _Float16* __restrict__ p) {
  const v8h lo = *(const v8h*)(p);
  const v8h hi = *(const v8h*)(p + 8);
  return __builtin_shufflevector(lo, hi, 0,1,2,3,4,5,6,7,8,9,10,11,12,13,14,15);
}

// f32 source A fragment (used only for softmaxed probabilities)
__device__ inline v16h load_a32(const float* __restrict__ rowp, int half) {
  v16h a;
  const float* p0 = rowp + half * 16;
  const float* p1 = rowp + half * 16 + 8;
#pragma unroll
  for (int i = 0; i < 8; ++i) {
    a[i]     = (_Float16)p0[i];
    a[8 + i] = (_Float16)p1[i];
  }
  return a;
}

#define WMMA_F16(A, B, C) \
  __builtin_amdgcn_wmma_f32_16x16x32_f16(false, (A), false, (B), (short)0, (C), false, false)

// ------------------------------------------------------------------
// f32 -> f16 elementwise convert (4 elems/thread; n % 1024 == 0)
// ------------------------------------------------------------------
__global__ __launch_bounds__(256) void cvt_f16(
    const float* __restrict__ in, _Float16* __restrict__ out, int n) {
  const int i = (blockIdx.x * 256 + threadIdx.x) * 4;
  if (i < n) {
    const float4 v = *(const float4*)(in + i);
    v8h dummy;  (void)dummy;
    _Float16 o0 = (_Float16)v.x, o1 = (_Float16)v.y;
    _Float16 o2 = (_Float16)v.z, o3 = (_Float16)v.w;
    typedef __attribute__((ext_vector_type(4))) _Float16 v4h;
    *(v4h*)(out + i) = (v4h){o0, o1, o2, o3};
  }
}

// ------------------------------------------------------------------
// Generic f16 GEMM: Y[M,N] = X[M,K] * W[N,K]^T + bias.
// Wave tile: 32 rows x 64 cols (2 M-frags x 4 N-frags).
// Block: 128 threads (4 waves stacked along M -> 128x64 block tile).
// Grid: (N/64, M/128).
// ------------------------------------------------------------------
template <bool RELU, bool W16, bool W32>
__global__ __launch_bounds__(128) void gemm16(
    const _Float16* __restrict__ X, const _Float16* __restrict__ W,
    const float* __restrict__ bias,
    float* __restrict__ Y32, _Float16* __restrict__ Y16,
    int M, int N, int K) {
  const int lane = threadIdx.x & 31;
  const int wave = threadIdx.x >> 5;
  const int half = lane >> 4;
  const int l16  = lane & 15;
  const int tileM = blockIdx.y * 128 + wave * 32;
  const int tileN = blockIdx.x * 64;

  v8f acc[2][4] = {};
  const _Float16* xrow0 = X + (size_t)(tileM + l16) * K;
  const _Float16* xrow1 = xrow0 + (size_t)16 * K;

  for (int k0 = 0; k0 < K; k0 += 32) {
    const v16h a0 = load_a16(xrow0 + k0, half);
    const v16h a1 = load_a16(xrow1 + k0, half);
#pragma unroll
    for (int c = 0; c < 4; ++c) {
      const _Float16* wp = W + (size_t)(tileN + c * 16 + l16) * K + k0 + half * 16;
      const v16h b = load_b16(wp);
      acc[0][c] = WMMA_F16(a0, b, acc[0][c]);
      acc[1][c] = WMMA_F16(a1, b, acc[1][c]);
    }
  }

#pragma unroll
  for (int r = 0; r < 2; ++r) {
#pragma unroll
    for (int c = 0; c < 4; ++c) {
      const int col = tileN + c * 16 + l16;
      const float bv = bias ? bias[col] : 0.0f;
#pragma unroll
      for (int j = 0; j < 8; ++j) {
        const int row = tileM + r * 16 + j + half * 8;
        float v = acc[r][c][j] + bv;
        if (RELU) v = v > 0.0f ? v : 0.0f;
        if (W32) Y32[(size_t)row * N + col] = v;
        if (W16) Y16[(size_t)row * N + col] = (_Float16)v;
      }
    }
  }
}

// ------------------------------------------------------------------
// scores[b,h,q,k] = scale * dot(Q[b,q,h,:], K[b,k,h,:])  (pure f16 loads)
// qkv16 packed [B,S,3E]; Q at +0, K at +E, head h at h*DH.
// Grid: (S/64, S/128, B*H). Block 128.
// ------------------------------------------------------------------
__global__ __launch_bounds__(128) void attn_scores16(
    const _Float16* __restrict__ qkv16, float* __restrict__ attn, float scale) {
  const int z = blockIdx.z;
  const int b = z >> 4;
  const int h = z & 15;
  const int lane = threadIdx.x & 31;
  const int wave = threadIdx.x >> 5;
  const int half = lane >> 4;
  const int l16  = lane & 15;
  const int tileM = blockIdx.y * 128 + wave * 32;
  const int tileN = blockIdx.x * 64;

  const _Float16* qbase = qkv16 + (size_t)(b * Ss) * (3 * E) + h * DH;
  const _Float16* kbase = qbase + E;

  v8f acc[2][4] = {};
  const _Float16* qr0 = qbase + (size_t)(tileM + l16) * (3 * E);
  const _Float16* qr1 = qr0 + (size_t)16 * (3 * E);

#pragma unroll
  for (int k0 = 0; k0 < DH; k0 += 32) {
    const v16h a0 = load_a16(qr0 + k0, half);
    const v16h a1 = load_a16(qr1 + k0, half);
#pragma unroll
    for (int c = 0; c < 4; ++c) {
      const _Float16* kp = kbase + (size_t)(tileN + c * 16 + l16) * (3 * E) + k0 + half * 16;
      const v16h bf = load_b16(kp);
      acc[0][c] = WMMA_F16(a0, bf, acc[0][c]);
      acc[1][c] = WMMA_F16(a1, bf, acc[1][c]);
    }
  }

  float* out = attn + (size_t)z * Ss * Ss;
#pragma unroll
  for (int r = 0; r < 2; ++r) {
#pragma unroll
    for (int c = 0; c < 4; ++c) {
      const int col = tileN + c * 16 + l16;
#pragma unroll
      for (int j = 0; j < 8; ++j) {
        const int row = tileM + r * 16 + j + half * 8;
        out[(size_t)row * Ss + col] = acc[r][c][j] * scale;
      }
    }
  }
}

// ------------------------------------------------------------------
// Softmax over last dim (1024) with additive mask, in place on f32.
// One 256-thread block per row; rows = B*H*S.
// ------------------------------------------------------------------
__global__ __launch_bounds__(256) void softmax_mask(
    float* __restrict__ attn, const float* __restrict__ mask) {
  __shared__ float sdata[256];
  const size_t row = blockIdx.x;
  const int q = (int)(row & (Ss - 1));
  float* p = attn + row * Ss;
  const float* m = mask + (size_t)q * Ss;
  const int t = threadIdx.x;

  float vals[4];
  float mx = -3.402823466e+38f;
#pragma unroll
  for (int i = 0; i < 4; ++i) {
    const int c = t + i * 256;
    vals[i] = p[c] + m[c];
    mx = fmaxf(mx, vals[i]);
  }
  sdata[t] = mx; __syncthreads();
  for (int s = 128; s > 0; s >>= 1) {
    if (t < s) sdata[t] = fmaxf(sdata[t], sdata[t + s]);
    __syncthreads();
  }
  mx = sdata[0]; __syncthreads();

  float sum = 0.0f;
#pragma unroll
  for (int i = 0; i < 4; ++i) {
    vals[i] = expf(vals[i] - mx);
    sum += vals[i];
  }
  sdata[t] = sum; __syncthreads();
  for (int s = 128; s > 0; s >>= 1) {
    if (t < s) sdata[t] += sdata[t + s];
    __syncthreads();
  }
  const float inv = 1.0f / sdata[0];
#pragma unroll
  for (int i = 0; i < 4; ++i) p[t + i * 256] = vals[i] * inv;
}

// ------------------------------------------------------------------
// V transpose: vT[(z*DH + d)*S + s] = qkv16[(b*S+s)*3E + 2E + h*DH + d]
// One block per (z,d) row; coalesced writes, L2-served strided reads.
// ------------------------------------------------------------------
__global__ __launch_bounds__(256) void transpose_v(
    const _Float16* __restrict__ qkv16, _Float16* __restrict__ vT) {
  const int r = blockIdx.x;           // z*DH + d
  const int z = r >> 6;
  const int d = r & 63;
  const int b = z >> 4;
  const int h = z & 15;
  const _Float16* src = qkv16 + (size_t)(b * Ss) * (3 * E) + 2 * E + h * DH + d;
  _Float16* dst = vT + (size_t)r * Ss;
  for (int s = threadIdx.x; s < Ss; s += 256)
    dst[s] = src[(size_t)s * (3 * E)];
}

// ------------------------------------------------------------------
// ctx16[b,q,h*DH+n] = sum_k P[b,h,q,k] * V[b,k,h,n]
// A: f32 probs (converted on load). B: vT f16 contiguous.
// Grid: (1, S/128, B*H). Block 128. Output written as f16.
// ------------------------------------------------------------------
__global__ __launch_bounds__(128) void attn_ctx16(
    const float* __restrict__ attn, const _Float16* __restrict__ vT,
    _Float16* __restrict__ ctxh) {
  const int z = blockIdx.z;
  const int b = z >> 4;
  const int h = z & 15;
  const int lane = threadIdx.x & 31;
  const int wave = threadIdx.x >> 5;
  const int half = lane >> 4;
  const int l16  = lane & 15;
  const int tileM = blockIdx.y * 128 + wave * 32;

  v8f acc[2][4] = {};
  const float* ar0 = attn + (size_t)z * Ss * Ss + (size_t)(tileM + l16) * Ss;
  const float* ar1 = ar0 + (size_t)16 * Ss;
  const _Float16* vtb = vT + (size_t)z * DH * Ss;   // [d][s]

  for (int k0 = 0; k0 < Ss; k0 += 32) {
    const v16h a0 = load_a32(ar0 + k0, half);
    const v16h a1 = load_a32(ar1 + k0, half);
#pragma unroll
    for (int c = 0; c < 4; ++c) {
      const int n = c * 16 + l16;
      const v16h bf = load_b16(vtb + (size_t)n * Ss + k0 + half * 16);
      acc[0][c] = WMMA_F16(a0, bf, acc[0][c]);
      acc[1][c] = WMMA_F16(a1, bf, acc[1][c]);
    }
  }

#pragma unroll
  for (int r = 0; r < 2; ++r) {
#pragma unroll
    for (int c = 0; c < 4; ++c) {
      const int n = c * 16 + l16;
#pragma unroll
      for (int j = 0; j < 8; ++j) {
        const int row = tileM + r * 16 + j + half * 8;
        ctxh[(size_t)(b * Ss + row) * E + h * DH + n] = (_Float16)acc[r][c][j];
      }
    }
  }
}

// ------------------------------------------------------------------
// y = LayerNorm(a + b) * g + beta; optional f16 copy of the output.
// One 256-thread block per row (E=1024).
// ------------------------------------------------------------------
template <bool W16>
__global__ __launch_bounds__(256) void add_layernorm(
    const float* __restrict__ a, const float* __restrict__ bres,
    const float* __restrict__ g, const float* __restrict__ beta,
    float* __restrict__ y32, _Float16* __restrict__ y16) {
  __shared__ float ssum[256];
  __shared__ float ssq[256];
  const size_t row = blockIdx.x;
  const int t = threadIdx.x;
  const float* pa = a + row * E;
  const float* pb = bres + row * E;

  float v[4];
  float s = 0.0f, sq = 0.0f;
#pragma unroll
  for (int i = 0; i < 4; ++i) {
    const int c = t + i * 256;
    v[i] = pa[c] + pb[c];
    s += v[i];
    sq += v[i] * v[i];
  }
  ssum[t] = s; ssq[t] = sq; __syncthreads();
  for (int st = 128; st > 0; st >>= 1) {
    if (t < st) { ssum[t] += ssum[t + st]; ssq[t] += ssq[t + st]; }
    __syncthreads();
  }
  const float mu   = ssum[0] * (1.0f / E);
  const float var  = ssq[0] * (1.0f / E) - mu * mu;
  const float rstd = rsqrtf(var + EPS);
  float* py = y32 + row * E;
#pragma unroll
  for (int i = 0; i < 4; ++i) {
    const int c = t + i * 256;
    const float o = (v[i] - mu) * rstd * g[c] + beta[c];
    py[c] = o;
    if (W16) y16[row * E + c] = (_Float16)o;
  }
}

// ------------------------------------------------------------------
extern "C" void kernel_launch(void* const* d_in, const int* in_sizes, int n_in,
                              void* d_out, int out_size, void* d_ws, size_t ws_size,
                              hipStream_t stream) {
  const float* src      = (const float*)d_in[0];
  const float* src_mask = (const float*)d_in[1];
  const float* w_qkv    = (const float*)d_in[2];
  const float* b_qkv    = (const float*)d_in[3];
  const float* w_o      = (const float*)d_in[4];
  const float* b_o      = (const float*)d_in[5];
  const float* w1       = (const float*)d_in[6];
  const float* b1       = (const float*)d_in[7];
  const float* w2       = (const float*)d_in[8];
  const float* b2       = (const float*)d_in[9];
  const float* ln1_g    = (const float*)d_in[10];
  const float* ln1_b    = (const float*)d_in[11];
  const float* ln2_g    = (const float*)d_in[12];
  const float* ln2_b    = (const float*)d_in[13];

  float* x_out    = (float*)d_out;                  // [B,S,E]
  float* attn_w   = x_out + (size_t)Bb * Ss * E;    // [B,H,S,S]

  // ---- workspace carve (bytes; lifetime-based reuse; total = 128 MiB) ----
  char* w = (char*)d_ws;
  _Float16* qkv16  = (_Float16*)(w);                 // 25,165,824 B  [B,S,3E]
  _Float16* src16  = (_Float16*)(w + 25165824);      //  8,388,608 B  (dead after QKV)
  _Float16* vT     = (_Float16*)(w + 33554432);      //  8,388,608 B  (dead after ctx)
  float*    x1_32  = (float*)   (w + 41943040);      // 16,777,216 B
  _Float16* hid16  = (_Float16*)(w + 58720256);      // 33,554,432 B  [B*S,F]
  float*    aout32 = (float*)   (w + 92274688);      // 16,777,216 B  (dead after LN1)
  _Float16* wq16   = (_Float16*)(w + 109051904);     //  6,291,456 B
  _Float16* wo16   = (_Float16*)(w + 115343360);     //  2,097,152 B
  _Float16* w116   = (_Float16*)(w + 117440512);     //  8,388,608 B
  _Float16* w216   = (_Float16*)(w + 125829120);     //  8,388,608 B
  _Float16* ctx16  = src16;                          // reuse
  _Float16* x1_16  = vT;                             // reuse
  float*    y2_32  = aout32;                         // reuse

  const int M = Bb * Ss;  // 4096
  const float scale = 1.0f / sqrtf((float)DH);

  // 0) one-time f16 conversions (weights + src)
  cvt_f16<<<(M * E) / 1024, 256, 0, stream>>>(src, src16, M * E);
  cvt_f16<<<(3 * E * E) / 1024, 256, 0, stream>>>(w_qkv, wq16, 3 * E * E);
  cvt_f16<<<(E * E) / 1024, 256, 0, stream>>>(w_o, wo16, E * E);
  cvt_f16<<<(Ff * E) / 1024, 256, 0, stream>>>(w1, w116, Ff * E);
  cvt_f16<<<(E * Ff) / 1024, 256, 0, stream>>>(w2, w216, E * Ff);

  // 1) QKV projection -> f16 qkv
  gemm16<false, true, false><<<dim3(3 * E / 64, M / 128), 128, 0, stream>>>(
      src16, wq16, b_qkv, nullptr, qkv16, M, 3 * E, E);

  // 2) transpose V for contiguous B-fragments in P·V
  transpose_v<<<Bb * Hh * DH, 256, 0, stream>>>(qkv16, vT);

  // 3) scores = scale * Q K^T -> f32 attn section of d_out
  attn_scores16<<<dim3(Ss / 64, Ss / 128, Bb * Hh), 128, 0, stream>>>(
      qkv16, attn_w, scale);

  // 4) softmax(+mask) in place (this IS the returned attn_weights)
  softmax_mask<<<Bb * Hh * Ss, 256, 0, stream>>>(attn_w, src_mask);

  // 5) ctx = P V -> f16
  attn_ctx16<<<dim3(1, Ss / 128, Bb * Hh), 128, 0, stream>>>(attn_w, vT, ctx16);

  // 6) out projection -> f32 (residual input)
  gemm16<false, false, true><<<dim3(E / 64, M / 128), 128, 0, stream>>>(
      ctx16, wo16, b_o, aout32, nullptr, M, E, E);

  // 7) x1 = LN(src + attn_out), f32 + f16 copies
  add_layernorm<true><<<M, 256, 0, stream>>>(src, aout32, ln1_g, ln1_b, x1_32, x1_16);

  // 8) hid = relu(x1 W1^T + b1) -> f16
  gemm16<true, true, false><<<dim3(Ff / 64, M / 128), 128, 0, stream>>>(
      x1_16, w116, b1, nullptr, hid16, M, Ff, E);

  // 9) y2 = hid W2^T + b2 -> f32
  gemm16<false, false, true><<<dim3(E / 64, M / 128), 128, 0, stream>>>(
      hid16, w216, b2, y2_32, nullptr, M, E, Ff);

  // 10) x = LN(x1 + y2) -> first output
  add_layernorm<false><<<M, 256, 0, stream>>>(x1_32, y2_32, ln2_g, ln2_b, x_out, nullptr);
}
